// GCNEdge_67817533604442
// MI455X (gfx1250) — compile-verified, matrix operand order checked by reading, not run on hardware
//
#include <hip/hip_runtime.h>
#include <math.h>

// Problem constants (match reference)
#define NNODES 30000
#define NEDGES 150000
#define IN_F   16
#define HID_F  32
#define CLS_F  10
#define KDIM   512   // IN_F * HID_F
#define EPB    128   // edges per block (8 waves x 16-edge M-tiles)

typedef __attribute__((ext_vector_type(16))) __bf16 v16bf;
typedef __attribute__((ext_vector_type(8)))  float  v8f;
typedef int v4i_b __attribute__((vector_size(16)));   // builtin's int4 type

// ---------------------------------------------------------------------------
// A/B fragment K-index map for V_WMMA_F32_16X16X32_BF16 (wave32), per ISA
// CDNA5 ISA "16-bit A-Matrix 16x32": lanes 0-15 hold K {0..7,16..23},
// lanes 16-31 hold K {8..15,24..31}; element e of the 16-element fragment:
//   v = e/2, p = e%2, klocal = (v<4?0:16) + 8*half + 2*(v&3) + p
// ---------------------------------------------------------------------------
__device__ __forceinline__ int frag_klocal(int e, int half) {
    int v = e >> 1, p = e & 1;
    return ((v < 4) ? 0 : 16) + 8 * half + 2 * (v & 3) + p;
}

// ---------------------------------------------------------------------------
// gfx1250 async global->LDS copy (ASYNCcnt path), 16 bytes per lane.
// Builtin signature on this toolchain (verified via round-2 diagnostic):
//   (int4 addrspace(1)*, int4 addrspace(3)*, imm offset, imm cpol)
// ---------------------------------------------------------------------------
__device__ __forceinline__ void async_copy_b128(const void* g, void* l) {
#if __has_builtin(__builtin_amdgcn_global_load_async_to_lds_b128)
    __builtin_amdgcn_global_load_async_to_lds_b128(
        (__attribute__((address_space(1))) v4i_b*)g,
        (__attribute__((address_space(3))) v4i_b*)l, 0, 0);
#else
    unsigned loff = (unsigned)(unsigned long long)
        (__attribute__((address_space(3))) char*)l;
    unsigned long long ga = (unsigned long long)(size_t)g;
    asm volatile("global_load_async_to_lds_b128 %0, %1, off"
                 :: "v"(loff), "v"(ga) : "memory");
#endif
}

__device__ __forceinline__ void wait_async0() {
#if __has_builtin(__builtin_amdgcn_s_wait_asynccnt)
    __builtin_amdgcn_s_wait_asynccnt(0);
#else
    asm volatile("s_wait_asynccnt 0x0" ::: "memory");
#endif
}

// ---------------------------------------------------------------------------
// Kernel 0: pack W2 [512,512] f32 (row k, col n) into bf16 fragment layout:
// element index = ((nt*16 + kt)*32 + lane)*16 + e  -> each lane's fragment is
// 32 contiguous bytes; each N-tile (nt) is one contiguous 16 KB block, which
// is exactly the unit the async staging loop DMAs into LDS.
// ---------------------------------------------------------------------------
__global__ void pack_w2_kernel(const float* __restrict__ W2,
                               __bf16* __restrict__ w2p) {
    int idx = blockIdx.x * 256 + threadIdx.x;        // 0 .. 512*512-1
    int e    = idx & 15;
    int lane = (idx >> 4) & 31;
    int kt   = (idx >> 9) & 15;
    int nt   = idx >> 13;
    int half = lane >> 4;
    int n    = nt * 16 + (lane & 15);
    int k    = kt * 32 + frag_klocal(e, half);
    w2p[idx] = (__bf16)W2[(size_t)k * KDIM + n];
}

// ---------------------------------------------------------------------------
// Kernel 1: in-degree counts (mean aggregation divisor; GCN deg = cnt+1)
// ---------------------------------------------------------------------------
__global__ void count_deg_kernel(const int* __restrict__ dst,
                                 float* __restrict__ cnt) {
    int e = blockIdx.x * 256 + threadIdx.x;
    if (e < NEDGES) atomicAdd(&cnt[dst[e]], 1.0f);
}

// ---------------------------------------------------------------------------
// Kernel 2: fused NNConv message kernel (the 78.6 GFLOP GEMM).
//   Per block: 128 edges; 8 waves = 8 independent 16-edge M-tiles, all
//   sharing one LDS-staged B tile per N-step (8x less L2/B traffic than
//   per-wave loads).  B tiles are double-buffered via async global->LDS
//   DMA (ASYNCcnt) overlapped with the 16-WMMA K-loop; B fragments are
//   additionally double-buffered in registers inside the K-loop so each
//   WMMA never waits on its own ds_loads.
//   A fragments (h = relu(a_e*W1+b1), bf16) are synthesized in registers
//   from the scalar edge attribute -> h never hits memory; theta stays in
//   the WMMA accumulator and is contracted with x[src] immediately, then
//   atomically scattered into summed[dst] -> theta never hits memory.
// ---------------------------------------------------------------------------
__global__ __launch_bounds__(256, 1)
void nnconv_msg_kernel(const float* __restrict__ x,
                       const float* __restrict__ eattr,
                       const float* __restrict__ W1,
                       const float* __restrict__ b1,
                       const float* __restrict__ b2,
                       const int*   __restrict__ src,
                       const int*   __restrict__ dst,
                       const __bf16* __restrict__ w2p,
                       float*       __restrict__ summed) {
    __shared__ float w1s[KDIM], b1s[KDIM], b2s[KDIM];
    __shared__ float attr_s[EPB];
    __shared__ float xs[EPB * IN_F];
    __shared__ int   src_s[EPB], dst_s[EPB];
    __shared__ v16bf bbuf[2][512];   // double-buffered 16 KB B tiles

    const int tid   = threadIdx.x;
    const int ebase = blockIdx.x * EPB;

    for (int i = tid; i < KDIM; i += 256) {
        w1s[i] = W1[i]; b1s[i] = b1[i]; b2s[i] = b2[i];
    }
    for (int i = tid; i < EPB; i += 256) {
        int e = ebase + i;
        int s = 0, d = 0; float a = 0.0f;
        if (e < NEDGES) { s = src[e]; d = dst[e]; a = eattr[e]; }
        src_s[i] = s; dst_s[i] = d; attr_s[i] = a;
    }
    __syncthreads();
    for (int i = tid; i < EPB * IN_F; i += 256) {
        xs[i] = x[(size_t)src_s[i >> 4] * IN_F + (i & 15)];
    }

    // kick off async stage of N-tile 0 while we build A fragments
    {
        const char* gsrc = (const char*)w2p + (size_t)tid * 16;
        char*       ldst = (char*)&bbuf[0][0] + (size_t)tid * 16;
        #pragma unroll
        for (int i = 0; i < 4; ++i)
            async_copy_b128(gsrc + i * 4096, ldst + i * 4096);
    }

    const int lane = tid & 31;
    const int w    = tid >> 5;    // this wave's 16-edge M-tile (0..7)
    const int half = lane >> 4;
    const int col  = lane & 15;

    // --- synthesize all 16 A fragments (K = 0..511) from scalar a_e ---
    const float ar = attr_s[w * 16 + col];   // A row M = lane%16
    v16bf afrag[16];
    #pragma unroll
    for (int kt = 0; kt < 16; ++kt) {
        v16bf a;
        #pragma unroll
        for (int e2 = 0; e2 < 16; ++e2) {
            int k = kt * 32 + frag_klocal(e2, half);
            float hv = fmaf(ar, w1s[k], b1s[k]);
            a[e2] = (__bf16)(hv > 0.0f ? hv : 0.0f);
        }
        afrag[kt] = a;
    }

    float msg[2][8];
    #pragma unroll
    for (int h2 = 0; h2 < 2; ++h2)
        #pragma unroll
        for (int v = 0; v < 8; ++v) msg[h2][v] = 0.0f;

    // --- 32 N-tiles; per tile: wait DMA, barrier, prefetch next tile,
    //     16-WMMA K-loop from LDS, fused x_src contraction ---
    #pragma unroll 1
    for (int j = 0; j < 32; ++j) {
        const int cur = j & 1;
        wait_async0();        // own stage-j chunks have landed in LDS
        __syncthreads();      // => everyone's chunks landed; prev reads done

        if (j + 1 < 32) {     // async-prefetch next 16 KB B tile
            const char* gsrc = (const char*)w2p
                             + (size_t)(j + 1) * 16384 + (size_t)tid * 16;
            char* ldst = (char*)&bbuf[cur ^ 1][0] + (size_t)tid * 16;
            #pragma unroll
            for (int i = 0; i < 4; ++i)
                async_copy_b128(gsrc + i * 4096, ldst + i * 4096);
        }

        // K-loop: register-double-buffered B fragments feeding WMMA
        v8f acc = 0.0f;
        v16bf bcur = bbuf[cur][lane];
        #pragma unroll
        for (int kt = 0; kt < 16; ++kt) {
            v16bf bnext = bcur;
            if (kt < 15) bnext = bbuf[cur][(kt + 1) * 32 + lane];
            acc = __builtin_amdgcn_wmma_f32_16x16x32_bf16(
                false, afrag[kt], false, bcur, (short)0, acc, false, false);
            bcur = bnext;
        }

        // theta[i*32+o]: i = j>>1, o = (j&1)*16 + col (this lane's column)
        const float b2v = b2s[j * 16 + col];
        const int   i   = j >> 1;
        const int   h2  = j & 1;
        #pragma unroll
        for (int v = 0; v < 8; ++v) {
            int m = v + 8 * half;             // D-layout: M = v + 8*(lane/16)
            float xv = xs[(w * 16 + m) * IN_F + i];
            msg[h2][v] = fmaf(xv, acc[v] + b2v, msg[h2][v]);
        }
    }

    // --- scatter msg into summed[dst] (both output halves) ---
    #pragma unroll
    for (int h2 = 0; h2 < 2; ++h2) {
        #pragma unroll
        for (int v = 0; v < 8; ++v) {
            int mloc = w * 16 + v + 8 * half;
            int e = ebase + mloc;
            if (e < NEDGES) {
                atomicAdd(&summed[(size_t)dst_s[mloc] * HID_F + h2 * 16 + col],
                          msg[h2][v]);
            }
        }
    }
}

// ---------------------------------------------------------------------------
// Kernel 3: node update: mean-aggr + root weight + bias + relu, then GCN
// feature transform xw = h1@Wg, dinv = rsqrt(deg+1), and out init with the
// self-loop term xw*dinv^2 + bg.  (~0.1 GFLOP total, VALU is fine.)
// ---------------------------------------------------------------------------
__global__ void node_update_kernel(const float* __restrict__ summed,
                                   const float* __restrict__ cnt,
                                   const float* __restrict__ x,
                                   const float* __restrict__ root,
                                   const float* __restrict__ bias1,
                                   const float* __restrict__ Wg,
                                   const float* __restrict__ bg,
                                   float* __restrict__ xw,
                                   float* __restrict__ dinv,
                                   float* __restrict__ outacc) {
    int n = blockIdx.x * 256 + threadIdx.x;
    if (n >= NNODES) return;
    float c   = cnt[n];
    float inv = 1.0f / fmaxf(c, 1.0f);
    float h1[HID_F];
    #pragma unroll
    for (int o = 0; o < HID_F; ++o)
        h1[o] = summed[(size_t)n * HID_F + o] * inv + bias1[o];
    #pragma unroll
    for (int i = 0; i < IN_F; ++i) {
        float xv = x[(size_t)n * IN_F + i];
        #pragma unroll
        for (int o = 0; o < HID_F; ++o)
            h1[o] = fmaf(xv, root[i * HID_F + o], h1[o]);
    }
    #pragma unroll
    for (int o = 0; o < HID_F; ++o) h1[o] = fmaxf(h1[o], 0.0f);

    float dv = rsqrtf(c + 1.0f);    // GCN deg = indeg + self loop (>=1)
    dinv[n] = dv;
    #pragma unroll
    for (int cc = 0; cc < CLS_F; ++cc) {
        float s = 0.0f;
        #pragma unroll
        for (int o = 0; o < HID_F; ++o)
            s = fmaf(h1[o], Wg[o * CLS_F + cc], s);
        xw[(size_t)n * CLS_F + cc]     = s;
        outacc[(size_t)n * CLS_F + cc] = fmaf(s, dv * dv, bg[cc]);
    }
}

// ---------------------------------------------------------------------------
// Kernel 4: GCN edge scatter: out[dst] += xw[src]*dinv[src]*dinv[dst]
// ---------------------------------------------------------------------------
__global__ void gcn_scatter_kernel(const int* __restrict__ src,
                                   const int* __restrict__ dst,
                                   const float* __restrict__ xw,
                                   const float* __restrict__ dinv,
                                   float* __restrict__ outacc) {
    int e = blockIdx.x * 256 + threadIdx.x;
    if (e >= NEDGES) return;
    int s = src[e], d = dst[e];
    float wgt = dinv[s] * dinv[d];
    #pragma unroll
    for (int c = 0; c < CLS_F; ++c)
        atomicAdd(&outacc[(size_t)d * CLS_F + c],
                  xw[(size_t)s * CLS_F + c] * wgt);
}

// ---------------------------------------------------------------------------
// Kernel 5: row-wise log_softmax
// ---------------------------------------------------------------------------
__global__ void log_softmax_kernel(const float* __restrict__ outacc,
                                   float* __restrict__ out) {
    int n = blockIdx.x * 256 + threadIdx.x;
    if (n >= NNODES) return;
    float v[CLS_F], m = -3.0e38f;
    #pragma unroll
    for (int c = 0; c < CLS_F; ++c) {
        v[c] = outacc[(size_t)n * CLS_F + c];
        m = fmaxf(m, v[c]);
    }
    float s = 0.0f;
    #pragma unroll
    for (int c = 0; c < CLS_F; ++c) s += expf(v[c] - m);
    float l = m + logf(s);
    #pragma unroll
    for (int c = 0; c < CLS_F; ++c)
        out[(size_t)n * CLS_F + c] = v[c] - l;
}

// ---------------------------------------------------------------------------
extern "C" void kernel_launch(void* const* d_in, const int* in_sizes, int n_in,
                              void* d_out, int out_size, void* d_ws,
                              size_t ws_size, hipStream_t stream) {
    const float* x     = (const float*)d_in[0];
    const float* eattr = (const float*)d_in[1];
    const float* W1    = (const float*)d_in[2];
    const float* b1    = (const float*)d_in[3];
    const float* W2    = (const float*)d_in[4];
    const float* b2    = (const float*)d_in[5];
    const float* root  = (const float*)d_in[6];
    const float* bias1 = (const float*)d_in[7];
    const float* Wg    = (const float*)d_in[8];
    const float* bg    = (const float*)d_in[9];
    const int*   eidx  = (const int*)d_in[10];
    const int* src = eidx;
    const int* dst = eidx + NEDGES;

    // workspace carve-out (~7 MB total)
    char* ws = (char*)d_ws;
    size_t off = 0;
    auto carve = [&](size_t bytes) -> char* {
        char* p = ws + off;
        off = (off + bytes + 255) & ~(size_t)255;
        return p;
    };
    __bf16* w2p    = (__bf16*)carve((size_t)KDIM * KDIM * sizeof(__bf16));
    float*  summed = (float*)carve((size_t)NNODES * HID_F * sizeof(float));
    float*  cnt    = (float*)carve((size_t)NNODES * sizeof(float));
    float*  xw     = (float*)carve((size_t)NNODES * CLS_F * sizeof(float));
    float*  dinv   = (float*)carve((size_t)NNODES * sizeof(float));
    float*  outacc = (float*)carve((size_t)NNODES * CLS_F * sizeof(float));

    (void)hipMemsetAsync(summed, 0, (size_t)NNODES * HID_F * sizeof(float),
                         stream);
    (void)hipMemsetAsync(cnt, 0, (size_t)NNODES * sizeof(float), stream);

    pack_w2_kernel<<<(KDIM * KDIM) / 256, 256, 0, stream>>>(W2, w2p);
    count_deg_kernel<<<(NEDGES + 255) / 256, 256, 0, stream>>>(dst, cnt);
    nnconv_msg_kernel<<<(NEDGES + EPB - 1) / EPB, 256, 0, stream>>>(
        x, eattr, W1, b1, b2, src, dst, w2p, summed);
    node_update_kernel<<<(NNODES + 255) / 256, 256, 0, stream>>>(
        summed, cnt, x, root, bias1, Wg, bg, xw, dinv, outacc);
    gcn_scatter_kernel<<<(NEDGES + 255) / 256, 256, 0, stream>>>(
        src, dst, xw, dinv, outacc);
    log_softmax_kernel<<<(NNODES + 255) / 256, 256, 0, stream>>>(
        outacc, (float*)d_out);
}
